// LSTM_83150566850652
// MI455X (gfx1250) — matrix operand order repaired and checked by reading, not compiled, not run on hardware
//
#include <hip/hip_runtime.h>

// ---------------------------------------------------------------------------
// Fused LSTM cell for MI455X (gfx1250, wave32, WMMA).
//   pre[b,g,o] = h@Wh[g]^T + x@Wx[g]^T + a@Wa[g]^T + bias[g]
//   i,f,o = sigmoid(pre[0..2]); g = tanh(pre[3])
//   c = f*c_prev + i*g ; h = o*tanh(c)
// GEMM via 3x-BF16 split-precision WMMA:  A*B ~= Ah*Bh + Ah*Bl + Al*Bh.
// Tiling: WG = 64M x 64N x 4 gates, 8 wave32; wave = 2 M-subtiles x 16N x 4g.
// ---------------------------------------------------------------------------

typedef __bf16 bf16_t;
typedef __attribute__((ext_vector_type(4)))  float  v4f;
typedef __attribute__((ext_vector_type(8)))  float  v8f;
typedef __attribute__((ext_vector_type(8)))  bf16_t v8bf;
typedef __attribute__((ext_vector_type(16))) bf16_t v16bf;

#define BDIM 8192
#define HDIM 1024
#define EDIM 300
#define ADIM 512

#define WMMA_BF16(A, Bm, C) \
    __builtin_amdgcn_wmma_f32_16x16x32_bf16(false, A, false, Bm, (short)0, C, false, false)

__device__ __forceinline__ float fast_sigmoid(float x) {
    return 1.0f / (1.0f + __expf(-x));
}
__device__ __forceinline__ float fast_tanh(float x) {
    return 2.0f / (1.0f + __expf(-2.0f * x)) - 1.0f;
}

__device__ __forceinline__ v16bf cat16(v8bf a, v8bf b) {
    v16bf r;
#pragma unroll
    for (int i = 0; i < 8; ++i) { r[i] = a[i]; r[i + 8] = b[i]; }
    return r;
}

// fp32 -> (bf16 hi, bf16 lo) split of 8 values
__device__ __forceinline__ void split8(v4f a, v4f b, v8bf& hi, v8bf& lo) {
#pragma unroll
    for (int j = 0; j < 4; ++j) {
        bf16_t h = (bf16_t)a[j];
        hi[j] = h;
        lo[j] = (bf16_t)(a[j] - (float)h);
    }
#pragma unroll
    for (int j = 0; j < 4; ++j) {
        bf16_t h = (bf16_t)b[j];
        hi[4 + j] = h;
        lo[4 + j] = (bf16_t)(b[j] - (float)h);
    }
}

// One K-chunk of 32: stage fp32->bf16 hi/lo tiles into LDS, then 24 WMMAs/wave.
// LD is the segment's K dimension (compile-time); IS_TAIL handles LD%32 != 0.
template <int LD, bool IS_TAIL>
__device__ __forceinline__ void lstm_chunk(
    const float* __restrict__ X, const float* __restrict__ W,
    int kbase, int b_base, int o_base, int tid, int lane, int wm, int wn,
    bf16_t sA[2][64][32], bf16_t sB[2][4][64][32], v8f acc[2][4])
{
    __syncthreads();   // previous chunk's fragment reads complete

    // ---- Stage X tile: 64 rows x 32 k, 8 floats/thread ----
    {
        const int xrow = tid >> 2;          // 0..63
        const int xk   = (tid & 3) * 8;     // 0,8,16,24
        const float* xp = X + (size_t)(b_base + xrow) * LD + kbase + xk;
        v4f x0, x1;
        if (!IS_TAIL) {
            x0 = *(const v4f*)xp;
            x1 = *(const v4f*)(xp + 4);
        } else {
#pragma unroll
            for (int j = 0; j < 4; ++j) {
                x0[j] = (kbase + xk + j     < LD) ? xp[j]     : 0.0f;
                x1[j] = (kbase + xk + 4 + j < LD) ? xp[4 + j] : 0.0f;
            }
        }
        v8bf hx, lx;
        split8(x0, x1, hx, lx);
        *(v8bf*)&sA[0][xrow][xk] = hx;
        *(v8bf*)&sA[1][xrow][xk] = lx;
    }

    // ---- Stage W tile: 4 gates x 64 rows x 32 k, one row/thread ----
    {
        const int g_ = tid >> 6;            // 0..3
        const int o_ = tid & 63;            // 0..63
        const float* wp = W + ((size_t)g_ * HDIM + (o_base + o_)) * LD + kbase;
#pragma unroll
        for (int q = 0; q < 4; ++q) {
            const int k0 = q * 8;
            v4f w0, w1;
            if (!IS_TAIL) {
                w0 = *(const v4f*)(wp + k0);
                w1 = *(const v4f*)(wp + k0 + 4);
            } else {
#pragma unroll
                for (int j = 0; j < 4; ++j) {
                    w0[j] = (kbase + k0 + j     < LD) ? wp[k0 + j]     : 0.0f;
                    w1[j] = (kbase + k0 + 4 + j < LD) ? wp[k0 + 4 + j] : 0.0f;
                }
            }
            v8bf hw, lw;
            split8(w0, w1, hw, lw);
            *(v8bf*)&sB[0][g_][o_][k0] = hw;
            *(v8bf*)&sB[1][g_][o_][k0] = lw;
        }
        if (!IS_TAIL && kbase + 32 < LD)
            __builtin_prefetch(wp + 32, 0, 1);   // global_prefetch_b8
    }

    __syncthreads();   // tiles visible

    // ---- Fragments (ISA 16-bit A 16x32 / B 32x16 striping) ----
    const int ak = (lane >> 4) * 8;     // A: K {ak..ak+7, ak+16..ak+23}
    const int bk = (lane >> 4) * 8;     // B: K {bk..bk+7, bk+16..bk+23}
    const int bcol = wn * 16 + (lane & 15);

    v16bf ah[2], al[2];
#pragma unroll
    for (int mt = 0; mt < 2; ++mt) {
        const int arow = wm * 32 + mt * 16 + (lane & 15);
        ah[mt] = cat16(*(const v8bf*)&sA[0][arow][ak],
                       *(const v8bf*)&sA[0][arow][ak + 16]);
        al[mt] = cat16(*(const v8bf*)&sA[1][arow][ak],
                       *(const v8bf*)&sA[1][arow][ak + 16]);
    }

#pragma unroll
    for (int g = 0; g < 4; ++g) {
        v16bf bh = cat16(*(const v8bf*)&sB[0][g][bcol][bk],
                         *(const v8bf*)&sB[0][g][bcol][bk + 16]);
        v16bf bl_ = cat16(*(const v8bf*)&sB[1][g][bcol][bk],
                          *(const v8bf*)&sB[1][g][bcol][bk + 16]);
#pragma unroll
        for (int mt = 0; mt < 2; ++mt) {
            acc[mt][g] = WMMA_BF16(ah[mt], bh,  acc[mt][g]);
            acc[mt][g] = WMMA_BF16(ah[mt], bl_, acc[mt][g]);
            acc[mt][g] = WMMA_BF16(al[mt], bh,  acc[mt][g]);
        }
    }
}

// One input segment (X [B,LD] against W [4,H,LD]) with compile-time LD.
template <int LD>
__device__ __forceinline__ void lstm_segment(
    const float* __restrict__ X, const float* __restrict__ W,
    int b_base, int o_base, int tid, int lane, int wm, int wn,
    bf16_t sA[2][64][32], bf16_t sB[2][4][64][32], v8f acc[2][4])
{
#pragma unroll 1
    for (int ch = 0; ch < LD / 32; ++ch)
        lstm_chunk<LD, false>(X, W, ch * 32, b_base, o_base, tid, lane, wm, wn,
                              sA, sB, acc);
    if (LD % 32)
        lstm_chunk<LD, true>(X, W, (LD / 32) * 32, b_base, o_base, tid, lane,
                             wm, wn, sA, sB, acc);
}

__global__ __launch_bounds__(256) void lstm_wmma_kernel(
    const float* __restrict__ h_prev, const float* __restrict__ c_prev,
    const float* __restrict__ emb,    const float* __restrict__ ctx,
    const float* __restrict__ Wh,     const float* __restrict__ Wx,
    const float* __restrict__ Wa,
    const float* __restrict__ bl,     const float* __restrict__ bl2,
    const float* __restrict__ bl3,    const float* __restrict__ bp,
    float* __restrict__ out_h,        float* __restrict__ out_c)
{
    __shared__ bf16_t sA[2][64][32];      //  8 KB  (hi/lo X tile)
    __shared__ bf16_t sB[2][4][64][32];   // 32 KB  (hi/lo W tile, 4 gates)

    const int tid  = threadIdx.x;
    const int lane = tid & 31;
    const int wid  = tid >> 5;
    const int wm   = wid >> 2;        // 0..1 : which 32-row half
    const int wn   = wid & 3;         // 0..3 : 16-col N sub-tile

    const int b_base = blockIdx.x * 64;   // batch tile
    const int o_base = blockIdx.y * 64;   // output-feature tile

    v8f acc[2][4] = {};   // [m-subtile][gate] 16x16 f32 accumulators

    lstm_segment<HDIM>(h_prev, Wh, b_base, o_base, tid, lane, wm, wn, sA, sB, acc);
    lstm_segment<EDIM>(emb,    Wx, b_base, o_base, tid, lane, wm, wn, sA, sB, acc);
    lstm_segment<ADIM>(ctx,    Wa, b_base, o_base, tid, lane, wm, wn, sA, sB, acc);

    // ------------------------- Epilogue -------------------------
    const int ocol = o_base + wn * 16 + (lane & 15);
    const float bias_i = bl[0 * HDIM + ocol] + bl2[0 * HDIM + ocol] +
                         bl3[0 * HDIM + ocol] + bp[0 * HDIM + ocol];
    const float bias_f = bl[1 * HDIM + ocol] + bl2[1 * HDIM + ocol] +
                         bl3[1 * HDIM + ocol] + bp[1 * HDIM + ocol];
    const float bias_o = bl[2 * HDIM + ocol] + bl2[2 * HDIM + ocol] +
                         bl3[2 * HDIM + ocol] + bp[2 * HDIM + ocol];
    const float bias_g = bl[3 * HDIM + ocol] + bl2[3 * HDIM + ocol] +
                         bl3[3 * HDIM + ocol] + bp[3 * HDIM + ocol];

#pragma unroll
    for (int mt = 0; mt < 2; ++mt) {
#pragma unroll
        for (int r = 0; r < 8; ++r) {
            // C/D layout: VGPR r, lanes 0-15 -> M=r ; lanes 16-31 -> M=r+8
            const int brow = b_base + wm * 32 + mt * 16 + r + ((lane >> 4) << 3);
            const size_t idx = (size_t)brow * HDIM + ocol;

            const float it = fast_sigmoid(acc[mt][0][r] + bias_i);
            const float ft = fast_sigmoid(acc[mt][1][r] + bias_f);
            const float ot = fast_sigmoid(acc[mt][2][r] + bias_o);
            const float gt = fast_tanh   (acc[mt][3][r] + bias_g);

            const float cp = c_prev[idx];
            const float ct = ft * cp + it * gt;
            const float ht = ot * fast_tanh(ct);

            out_h[idx] = ht;
            out_c[idx] = ct;
        }
    }
}

extern "C" void kernel_launch(void* const* d_in, const int* in_sizes, int n_in,
                              void* d_out, int out_size, void* d_ws, size_t ws_size,
                              hipStream_t stream) {
    const float* h_prev = (const float*)d_in[0];
    const float* c_prev = (const float*)d_in[1];
    const float* emb    = (const float*)d_in[2];
    const float* ctx    = (const float*)d_in[3];
    const float* Wh     = (const float*)d_in[4];
    const float* Wx     = (const float*)d_in[5];
    const float* Wa     = (const float*)d_in[6];
    const float* bl     = (const float*)d_in[7];
    const float* bl2    = (const float*)d_in[8];
    const float* bl3    = (const float*)d_in[9];
    const float* bp     = (const float*)d_in[10];

    float* out_h = (float*)d_out;                         // h_t first
    float* out_c = (float*)d_out + (size_t)BDIM * HDIM;   // then c_t

    dim3 grid(BDIM / 64, HDIM / 64, 1);   // 128 x 16 workgroups
    dim3 block(256, 1, 1);                // 8 wave32
    lstm_wmma_kernel<<<grid, block, 0, stream>>>(
        h_prev, c_prev, emb, ctx, Wh, Wx, Wa, bl, bl2, bl3, bp, out_h, out_c);
}
